// CovFunction_14628658610178
// MI455X (gfx1250) — matrix operand order repaired and checked by reading, not compiled, not run on hardware
//
#include <hip/hip_runtime.h>

typedef __attribute__((ext_vector_type(16))) _Float16 v16h;
typedef __attribute__((ext_vector_type(8)))  _Float16 v8h;
typedef __attribute__((ext_vector_type(8)))  float    v8f;
typedef __attribute__((ext_vector_type(4)))  unsigned int u32x4;
typedef __attribute__((ext_vector_type(8)))  int          i32x8;
typedef __attribute__((ext_vector_type(4)))  int          i32x4;

#define N_ROWS 8192
#define M_ROWS 8192
#define D_K    512

// ---------------------------------------------------------------------------
// Preprocess: Xs = X * exp(-w); split fp32 -> f16 hi + f16 residual lo;
// row squared norms in fp32. One wave32 per row (D=512 -> 16 elems/lane).
// ---------------------------------------------------------------------------
__global__ __launch_bounds__(256)
void prep_kernel(const float* __restrict__ X,
                 const float* __restrict__ w,
                 _Float16* __restrict__ hi,
                 _Float16* __restrict__ lo,
                 float* __restrict__ nrm2)
{
    const int wave = threadIdx.x >> 5;
    const int lane = threadIdx.x & 31;
    const int row  = blockIdx.x * 8 + wave;

    const float* xr = X + (size_t)row * D_K;
    _Float16* hr = hi + (size_t)row * D_K;
    _Float16* lr = lo + (size_t)row * D_K;

    float acc = 0.0f;
    #pragma unroll 4
    for (int j = lane; j < D_K; j += 32) {
        float s = __expf(-w[j]);            // inverse lengthscale
        float x = xr[j] * s;
        _Float16 h = (_Float16)x;           // hi part
        float rem   = x - (float)h;         // residual
        hr[j] = h;
        lr[j] = (_Float16)rem;
        acc = fmaf(x, x, acc);
    }
    #pragma unroll
    for (int off = 16; off > 0; off >>= 1)
        acc += __shfl_xor(acc, off, 32);
    if (lane == 0) nrm2[row] = acc;
}

// ---------------------------------------------------------------------------
// TDM panel load: 128 rows x 32 K-halves (8 KB) from a row-major [rows][512]
// f16 matrix into LDS, via a 2-D Tensor DMA Descriptor (ISA 08 §8).
//   group0: count=1 | lds_addr | global_addr(tile start) | type=2
//   group1: data_size=2B, tensor_dim0=512, tensor_dim1=8192,
//           tile_dim0=32, tile_dim1=128, tensor_dim0_stride=512
//   groups 2/3: zero (2-D tensor)
// ---------------------------------------------------------------------------
__device__ __forceinline__ void tdm_load_panel(unsigned long long gaddr,
                                               unsigned lds_off)
{
    u32x4 g0;
    g0[0] = 1u;                                   // count=1, user descriptor
    g0[1] = lds_off;                              // LDS byte address
    g0[2] = (unsigned)(gaddr & 0xFFFFFFFFu);      // global_addr[31:0]
    g0[3] = (unsigned)((gaddr >> 32) & 0x1FFFFFFu)// global_addr[56:32]
          | (2u << 30);                           // type=2 ("image")
    i32x8 g1;
    g1[0] = (int)(1u << 16);                      // wg_mask=0, data_size=1 (2B)
    g1[1] = (int)(512u << 16);                    // tensor_dim0[15:0]=512
    g1[2] = (int)(8192u << 16);                   // dim0 hi=0, tensor_dim1 lo=8192
    g1[3] = (int)(32u << 16);                     // dim1 hi=0, tile_dim0=32
    g1[4] = 128;                                  // tile_dim1=128, tile_dim2=0
    g1[5] = 512;                                  // tensor_dim0_stride=512
    g1[6] = 0;
    g1[7] = 0;
    i32x4 gz4 = {0, 0, 0, 0};
    i32x8 gz8 = {0, 0, 0, 0, 0, 0, 0, 0};
    __builtin_amdgcn_tensor_load_to_lds(g0, g1, gz4, gz4, gz8, 0);
}

// ---------------------------------------------------------------------------
// Main kernel: split-f16 WMMA GEMM (Ahi*Bhi + Ahi*Blo + Alo*Bhi, f32 acc),
// operand panels staged LDS<-HBM/L2 by the Tensor Data Mover (double buffer),
// fused RBF epilogue  out = exp(2*sn - 0.5*max(u2+v2-2*dot, 0)).
// Block: 256 threads = 8 waves; 128x128 tile; each wave: 64x32.
// LDS: 2 bufs x 4 panels x 128x32 f16 = 64 KB (of 320 KB/WGP).
// ---------------------------------------------------------------------------
__global__ __launch_bounds__(256)
void rbf_wmma_kernel(const _Float16* __restrict__ Uhi, const _Float16* __restrict__ Ulo,
                     const _Float16* __restrict__ Vhi, const _Float16* __restrict__ Vlo,
                     const float* __restrict__ u2,  const float* __restrict__ v2,
                     const float* __restrict__ snp, float* __restrict__ out)
{
    __shared__ _Float16 smem[2][4][128][32];      // [buf][Ahi,Alo,Bhi,Blo][row][k]

    const int lane   = threadIdx.x & 31;
    const int wave   = threadIdx.x >> 5;
    const int wm     = wave & 1;                  // 2 waves along M
    const int wn     = wave >> 1;                 // 4 waves along N
    const int m0     = blockIdx.y * 128;
    const int n0     = blockIdx.x * 128;
    const int lm     = lane & 15;
    const bool hiHalf = (lane >= 16);

    // Global base addresses of the four 128-row panels for this block
    unsigned long long gbase[4];
    gbase[0] = (unsigned long long)(uintptr_t)(Uhi + (size_t)m0 * D_K);
    gbase[1] = (unsigned long long)(uintptr_t)(Ulo + (size_t)m0 * D_K);
    gbase[2] = (unsigned long long)(uintptr_t)(Vhi + (size_t)n0 * D_K);
    gbase[3] = (unsigned long long)(uintptr_t)(Vlo + (size_t)n0 * D_K);

    // Lane->K mapping used here: k = 16*(lane/16) + elem (wave32, 16-bit A 16x32 / B 32x16)
    const int aC0 = hiHalf ? 16 : 0;              // A: v16h elems 0..7 (contiguous K)
    const int aC1 = aC0 + 8;                      // A: v16h elems 8..15
    const int bC0 = hiHalf ? 16 : 0;              // B: contiguous 16 K-values
    const int aRowBase = wm * 64 + lm;            // + mt*16
    const int bRowBase = wn * 32 + lm;            // + nt*16

    // Prologue: DMA K-chunk 0 into buffer 0
    if (wave == 0) {
        #pragma unroll
        for (int p = 0; p < 4; ++p)
            tdm_load_panel(gbase[p], (unsigned)(uintptr_t)&smem[0][p][0][0]);
        __builtin_amdgcn_s_wait_tensorcnt(0);
    }
    __syncthreads();

    v8f acc[4][2] = {};
    int cur = 0;

    for (int kc = 0; kc < D_K; kc += 32) {
        // Kick off DMA of the next K-chunk into the other buffer
        if (kc + 32 < D_K && wave == 0) {
            #pragma unroll
            for (int p = 0; p < 4; ++p)
                tdm_load_panel(gbase[p] + (size_t)(kc + 32) * sizeof(_Float16),
                               (unsigned)(uintptr_t)&smem[cur ^ 1][p][0][0]);
        }

        const _Float16* sAh = &smem[cur][0][0][0];
        const _Float16* sAl = &smem[cur][1][0][0];
        const _Float16* sBh = &smem[cur][2][0][0];
        const _Float16* sBl = &smem[cur][3][0][0];

        v16h Ah[4], Al[4], Bh[2], Bl[2];
        #pragma unroll
        for (int mt = 0; mt < 4; ++mt) {
            const int ro = (aRowBase + mt * 16) * 32;
            v8h h0 = *(const v8h*)(sAh + ro + aC0);
            v8h h1 = *(const v8h*)(sAh + ro + aC1);
            v8h l0 = *(const v8h*)(sAl + ro + aC0);
            v8h l1 = *(const v8h*)(sAl + ro + aC1);
            #pragma unroll
            for (int i = 0; i < 8; ++i) {
                Ah[mt][i] = h0[i]; Ah[mt][8 + i] = h1[i];
                Al[mt][i] = l0[i]; Al[mt][8 + i] = l1[i];
            }
        }
        #pragma unroll
        for (int nt = 0; nt < 2; ++nt) {
            const int ro = (bRowBase + nt * 16) * 32;
            Bh[nt] = *(const v16h*)(sBh + ro + bC0);
            Bl[nt] = *(const v16h*)(sBl + ro + bC0);
        }

        // Split-precision product: hi*hi + hi*lo + lo*hi (lo*lo negligible)
        #pragma unroll
        for (int mt = 0; mt < 4; ++mt) {
            #pragma unroll
            for (int nt = 0; nt < 2; ++nt) {
                acc[mt][nt] = __builtin_amdgcn_wmma_f32_16x16x32_f16(
                    false, Ah[mt], false, Bh[nt], (short)0, acc[mt][nt], false, false);
                acc[mt][nt] = __builtin_amdgcn_wmma_f32_16x16x32_f16(
                    false, Ah[mt], false, Bl[nt], (short)0, acc[mt][nt], false, false);
                acc[mt][nt] = __builtin_amdgcn_wmma_f32_16x16x32_f16(
                    false, Al[mt], false, Bh[nt], (short)0, acc[mt][nt], false, false);
            }
        }

        // Next buffer must be resident before anyone reads it; barrier also
        // guarantees all waves are done with the current buffer.
        if (wave == 0) __builtin_amdgcn_s_wait_tensorcnt(0);
        __syncthreads();
        cur ^= 1;
    }

    // Fused RBF epilogue. C/D layout: VGPR r, lane l -> M = r + (l>=16 ? 8:0),
    // N = l%16 (half-wave writes 64B contiguous rows).
    const float two_sn = 2.0f * snp[0];
    #pragma unroll
    for (int mt = 0; mt < 4; ++mt) {
        const int mbase = m0 + wm * 64 + mt * 16 + (hiHalf ? 8 : 0);
        #pragma unroll
        for (int nt = 0; nt < 2; ++nt) {
            const int n  = n0 + wn * 32 + nt * 16 + lm;
            const float vn = v2[n];
            #pragma unroll
            for (int r = 0; r < 8; ++r) {
                const int m = mbase + r;
                float sq = fmaf(-2.0f, acc[mt][nt][r], u2[m] + vn);
                sq = fmaxf(sq, 0.0f);
                out[(size_t)m * M_ROWS + n] = __expf(fmaf(-0.5f, sq, two_sn));
            }
        }
    }
}

// ---------------------------------------------------------------------------
extern "C" void kernel_launch(void* const* d_in, const int* in_sizes, int n_in,
                              void* d_out, int out_size, void* d_ws, size_t ws_size,
                              hipStream_t stream) {
    (void)in_sizes; (void)n_in; (void)out_size; (void)ws_size;
    const float* U  = (const float*)d_in[0];   // [N,D]
    const float* V  = (const float*)d_in[1];   // [M,D]
    const float* w  = (const float*)d_in[2];   // [D,1]
    const float* sn = (const float*)d_in[3];   // scalar
    float* out = (float*)d_out;

    char* ws = (char*)d_ws;
    const size_t halfBytes = (size_t)N_ROWS * D_K * sizeof(_Float16); // 8 MiB
    _Float16* Uhi = (_Float16*)(ws);
    _Float16* Ulo = (_Float16*)(ws + 1 * halfBytes);
    _Float16* Vhi = (_Float16*)(ws + 2 * halfBytes);
    _Float16* Vlo = (_Float16*)(ws + 3 * halfBytes);
    float*    u2  = (float*)   (ws + 4 * halfBytes);
    float*    v2  = u2 + N_ROWS;

    prep_kernel<<<N_ROWS / 8, 256, 0, stream>>>(U, w, Uhi, Ulo, u2);
    prep_kernel<<<M_ROWS / 8, 256, 0, stream>>>(V, w, Vhi, Vlo, v2);

    dim3 grid(M_ROWS / 128, N_ROWS / 128);   // 64 x 64 workgroups
    rbf_wmma_kernel<<<grid, 256, 0, stream>>>(Uhi, Ulo, Vhi, Vlo, u2, v2, sn, out);
}
